// EncoderLayer_9070970929221
// MI455X (gfx1250) — matrix operand order repaired and checked by reading, not compiled, not run on hardware
//
#include <hip/hip_runtime.h>
#include <cstdint>
#include <cstddef>

#define DEVI __device__ __forceinline__

typedef __attribute__((ext_vector_type(8)))  __bf16 v8bf;
typedef __attribute__((ext_vector_type(16))) __bf16 v16bf;
typedef __attribute__((ext_vector_type(8)))  float  v8f;

constexpr int BB    = 4;
constexpr int SS    = 2048;
constexpr int DM    = 1024;
constexpr int DFF   = 4096;
constexpr int NH    = 16;
constexpr int DK    = 64;
constexpr int MROWS = BB * SS; // 8192

// Concatenate two 16-byte LDS/global reads into one 16x bf16 WMMA fragment.
DEVI v16bf ld16(const __bf16* lo, const __bf16* hi) {
  v8bf a = *(const v8bf*)lo;
  v8bf b = *(const v8bf*)hi;
  return __builtin_shufflevector(a, b, 0,1,2,3,4,5,6,7,8,9,10,11,12,13,14,15);
}

DEVI v8f wmma_bf16(v16bf a, v16bf b, v8f c) {
  // D = A(16x32) * B(32x16) + C, f32 accumulate
  return __builtin_amdgcn_wmma_f32_16x16x32_bf16(false, a, false, b, (short)0, c,
                                                 false, false);
}

// ---------------------------------------------------------------------------
// Weight convert + transpose: W[K][N] f32  ->  WT[N][K] bf16
// ---------------------------------------------------------------------------
__global__ __launch_bounds__(256)
void transpose_f32_to_bf16(const float* __restrict__ W, __bf16* __restrict__ WT,
                           int K, int N)
{
  __shared__ float tile[32][33];
  const int n0 = blockIdx.x * 32, k0 = blockIdx.y * 32;
  #pragma unroll
  for (int j = 0; j < 4; ++j) {
    const int kk = threadIdx.y + j * 8;
    tile[kk][threadIdx.x] = W[(size_t)(k0 + kk) * N + n0 + threadIdx.x];
  }
  __syncthreads();
  #pragma unroll
  for (int j = 0; j < 4; ++j) {
    const int nn = threadIdx.y + j * 8;
    WT[(size_t)(n0 + nn) * K + k0 + threadIdx.x] = (__bf16)tile[threadIdx.x][nn];
  }
}

// ---------------------------------------------------------------------------
// LayerNorm over D=1024, f32 in -> bf16 out. One block (256 thr) per row.
// ---------------------------------------------------------------------------
__global__ __launch_bounds__(256)
void layernorm_to_bf16(const float* __restrict__ x, const float* __restrict__ g,
                       const float* __restrict__ bta, __bf16* __restrict__ out)
{
  const int row = blockIdx.x;
  const int tid = threadIdx.x;
  const float4 xv = *(const float4*)(x + (size_t)row * DM + tid * 4);
  float s = xv.x + xv.y + xv.z + xv.w;
  float q = xv.x * xv.x + xv.y * xv.y + xv.z * xv.z + xv.w * xv.w;
  #pragma unroll
  for (int d = 16; d >= 1; d >>= 1) {
    s += __shfl_down(s, d, 32);
    q += __shfl_down(q, d, 32);
  }
  __shared__ float rs[8], rq[8];
  if ((tid & 31) == 0) { rs[tid >> 5] = s; rq[tid >> 5] = q; }
  __syncthreads();
  float ts = 0.f, tq = 0.f;
  #pragma unroll
  for (int i = 0; i < 8; ++i) { ts += rs[i]; tq += rq[i]; }
  const float mean = ts * (1.0f / DM);
  const float var  = tq * (1.0f / DM) - mean * mean;
  const float rstd = rsqrtf(var + 1e-6f);
  const float4 gv = *(const float4*)(g   + tid * 4);
  const float4 bv = *(const float4*)(bta + tid * 4);
  __bf16* o = out + (size_t)row * DM + tid * 4;
  o[0] = (__bf16)((xv.x - mean) * rstd * gv.x + bv.x);
  o[1] = (__bf16)((xv.y - mean) * rstd * gv.y + bv.y);
  o[2] = (__bf16)((xv.z - mean) * rstd * gv.z + bv.z);
  o[3] = (__bf16)((xv.w - mean) * rstd * gv.w + bv.w);
}

// ---------------------------------------------------------------------------
// bf16 WMMA GEMM:  C = A(MxK) * BT(NxK)^T + bias  [+ residual] [relu]
// Block tile 128x256, 256 threads = 8 waves (2x4), wave tile 64x64.
// Register-staged double-buffered LDS pipeline: one barrier / K-step,
// global loads for step k+1 overlap WMMAs of step k. All 8 fragments
// (4 A + 4 B) preloaded so the 16 WMMAs issue with staged dscnt waits.
// mode 0: scatter bf16 out to [b][h][s][64]   (QKV)
// mode 1: f32 out = val + residual            (proj / FFN2)
// mode 2: bf16 row-major out = relu(val)      (FFN1)
// ---------------------------------------------------------------------------
__global__ __launch_bounds__(256)
void gemm_bf16_wmma(const __bf16* __restrict__ A, const __bf16* __restrict__ BT,
                    const float* __restrict__ bias, const float* __restrict__ resid,
                    float* __restrict__ outF, __bf16* __restrict__ outB,
                    int M, int N, int K, int mode)
{
  __shared__ __bf16 Asl[2][128 * 40];   // 128 rows x 32 k, padded stride 40
  __shared__ __bf16 Bsl[2][256 * 40];   // 256 n-rows x 32 k, padded stride 40
  const int tid  = threadIdx.x;
  const int lane = tid & 31;
  const int wave = tid >> 5;
  const int ll = lane & 15, lh = lane >> 4;
  const int wm = wave & 1,  wn = wave >> 1;       // waves: 2 (M) x 4 (N)
  const int m0w = wm * 64,  n0w = wn * 64;
  const int mBlk = blockIdx.y * 128, nBlk = blockIdx.x * 256;

  const int rowA  = tid >> 1;
  const int koffA = (tid & 1) * 16;
  const __bf16* gA = A  + (size_t)(mBlk + rowA) * K + koffA;
  const __bf16* gB = BT + (size_t)(nBlk + tid) * K;

  const int kb  = lh * 8;   // A frag: K 0..7 / 8..15 per lane half
  const int kb2 = lh * 8;   // B frag: K 0..7 / 8..15 per lane half (same as A)

  v8f acc[4][4] = {};

  // staging registers (distinct, so all 6 loads stay in flight together)
  float4 ra0, ra1, rb0, rb1, rb2, rb3;
  auto load_regs = [&](int k0) {
    const float4* ga4 = (const float4*)(gA + k0);
    ra0 = ga4[0]; ra1 = ga4[1];
    const float4* gb4 = (const float4*)(gB + k0);
    rb0 = gb4[0]; rb1 = gb4[1]; rb2 = gb4[2]; rb3 = gb4[3];
    if (k0 + 32 < K) {
      __builtin_prefetch(gA + k0 + 32, 0, 3);
      __builtin_prefetch(gB + k0 + 32, 0, 3);
    }
  };
  auto store_lds = [&](int buf) {
    float4* sa = (float4*)&Asl[buf][rowA * 40 + koffA];
    sa[0] = ra0; sa[1] = ra1;
    float4* sb = (float4*)&Bsl[buf][tid * 40];
    sb[0] = rb0; sb[1] = rb1; sb[2] = rb2; sb[3] = rb3;
  };

  load_regs(0);
  store_lds(0);

  const int nk = K >> 5;
  for (int ki = 0; ki < nk; ++ki) {
    __syncthreads();
    const int cur = ki & 1;
    const bool more = (ki + 1) < nk;
    if (more) load_regs((ki + 1) * 32);

    // preload ALL fragments so the 16 WMMAs drain with staged waits only
    v16bf bfrag[4], afrag[4];
    #pragma unroll
    for (int nt = 0; nt < 4; ++nt) {
      const __bf16* p = &Bsl[cur][(n0w + nt * 16 + ll) * 40 + kb2];
      bfrag[nt] = ld16(p, p + 16);
    }
    #pragma unroll
    for (int mi = 0; mi < 4; ++mi) {
      const __bf16* p = &Asl[cur][(m0w + mi * 16 + ll) * 40 + kb];
      afrag[mi] = ld16(p, p + 16);
    }
    #pragma unroll
    for (int mi = 0; mi < 4; ++mi)
      #pragma unroll
      for (int nt = 0; nt < 4; ++nt)
        acc[mi][nt] = wmma_bf16(afrag[mi], bfrag[nt], acc[mi][nt]);

    if (more) store_lds(cur ^ 1);
  }

  // Epilogue. D layout: lane ll = column, row = i + 8*lh within each 16x16 tile.
  #pragma unroll
  for (int mi = 0; mi < 4; ++mi) {
    #pragma unroll
    for (int nt = 0; nt < 4; ++nt) {
      const int col = nBlk + n0w + nt * 16 + ll;
      const float bval = bias[col];
      const int rbase = mBlk + m0w + mi * 16 + lh * 8;
      #pragma unroll
      for (int i = 0; i < 8; ++i) {
        const int r = rbase + i;
        const float val = acc[mi][nt][i] + bval;
        if (mode == 0) {
          const int b = r >> 11, s = r & (SS - 1);
          const int h = col >> 6, d = col & (DK - 1);
          outB[(((size_t)(b * NH + h)) * SS + s) * DK + d] = (__bf16)val;
        } else if (mode == 2) {
          outB[(size_t)r * N + col] = (__bf16)fmaxf(val, 0.0f);
        } else {
          outF[(size_t)r * N + col] = val + resid[(size_t)r * N + col];
        }
      }
    }
  }
}

// ---------------------------------------------------------------------------
// Flash attention: grid.y = b*NH+h, grid.x = query block of 128.
// 8 waves x 16 queries; K/V chunks of 32 keys double-buffered in LDS.
// ---------------------------------------------------------------------------
__global__ __launch_bounds__(256)
void flash_attn_wmma(const __bf16* __restrict__ Q, const __bf16* __restrict__ Kd,
                     const __bf16* __restrict__ Vd, const int* __restrict__ mask,
                     __bf16* __restrict__ ctx)
{
  constexpr float L2E = 1.4426950408889634f;
  __shared__ __bf16 Kt[2][32 * 72];   // [key][64d], padded stride 72
  __shared__ __bf16 Vt[2][64 * 40];   // [d][32keys] (transposed), stride 40
  __shared__ __bf16 Pb[8][16 * 40];   // per-wave P scratch, stride 40

  const int bh = blockIdx.y;
  const int b  = bh >> 4, h = bh & 15;
  const int tid = threadIdx.x, lane = tid & 31, wave = tid >> 5;
  const int ll = lane & 15, lh = lane >> 4;
  const int q0 = blockIdx.x * 128 + wave * 16;

  // Q fragments (A-matrix layout: lane = row, K pairs split across lane halves)
  const __bf16* qrow = Q + ((size_t)bh * SS + q0 + ll) * DK;
  const int kb = lh * 8;
  const v16bf aq0 = ld16(qrow + kb,      qrow + kb + 16);        // d 0..31
  const v16bf aq1 = ld16(qrow + 32 + kb, qrow + 32 + kb + 16);   // d 32..63

  float mrow[8], lrow[8];
  #pragma unroll
  for (int i = 0; i < 8; ++i) { mrow[i] = -3.0e38f; lrow[i] = 0.0f; }
  v8f acc[4] = {};

  const int krow = tid >> 3, doff = (tid & 7) * 8;
  const int kb2 = lh * 8;

  float4 kreg, vreg;
  auto load_kv = [&](int key0) {
    kreg = *(const float4*)(Kd + ((size_t)bh * SS + key0 + krow) * DK + doff);
    vreg = *(const float4*)(Vd + ((size_t)bh * SS + key0 + krow) * DK + doff);
  };
  auto store_kv = [&](int buf) {
    *(float4*)&Kt[buf][krow * 72 + doff] = kreg;
    __bf16 vtmp[8];
    *(float4*)vtmp = vreg;
    #pragma unroll
    for (int j = 0; j < 8; ++j) Vt[buf][(doff + j) * 40 + krow] = vtmp[j];
  };

  load_kv(0);
  store_kv(0);

  constexpr int NCHUNK = SS / 32;
  for (int c = 0; c < NCHUNK; ++c) {
    __syncthreads();
    const int cur = c & 1;
    const bool more = (c + 1) < NCHUNK;
    if (more) load_kv((c + 1) * 32);
    const int key0 = c * 32;

    float sv[2][8];
    #pragma unroll
    for (int nt = 0; nt < 2; ++nt) {
      const __bf16* kp = &Kt[cur][(nt * 16 + ll) * 72 + kb2];
      const v16bf bk0 = ld16(kp,      kp + 16);  // d 0..31
      const v16bf bk1 = ld16(kp + 32, kp + 48);  // d 32..63
      v8f s = {};
      s = wmma_bf16(aq0, bk0, s);
      s = wmma_bf16(aq1, bk1, s);
      const int mv = mask[b * SS + key0 + nt * 16 + ll]; // per column
      #pragma unroll
      for (int i = 0; i < 8; ++i)
        sv[nt][i] = (mv != 0) ? s[i] * 0.125f : -1.0e9f;
    }

    // online softmax: rows live at i + 8*lh; reduce across 16-lane halves
    float pv[2][8], alpha[8];
    #pragma unroll
    for (int i = 0; i < 8; ++i) {
      float cm = fmaxf(sv[0][i], sv[1][i]);
      #pragma unroll
      for (int d = 8; d >= 1; d >>= 1) cm = fmaxf(cm, __shfl_xor(cm, d, 32));
      const float mn = fmaxf(mrow[i], cm);
      alpha[i] = exp2f((mrow[i] - mn) * L2E);
      pv[0][i] = exp2f((sv[0][i] - mn) * L2E);
      pv[1][i] = exp2f((sv[1][i] - mn) * L2E);
      float rsum = pv[0][i] + pv[1][i];
      #pragma unroll
      for (int d = 8; d >= 1; d >>= 1) rsum += __shfl_xor(rsum, d, 32);
      lrow[i] = lrow[i] * alpha[i] + rsum;
      mrow[i] = mn;
    }
    #pragma unroll
    for (int t = 0; t < 4; ++t)
      #pragma unroll
      for (int i = 0; i < 8; ++i) acc[t][i] *= alpha[i];

    // D-layout -> A-fragment layout via wave-private LDS
    __bf16* Pw = &Pb[wave][0];
    #pragma unroll
    for (int nt = 0; nt < 2; ++nt)
      #pragma unroll
      for (int i = 0; i < 8; ++i)
        Pw[(i + 8 * lh) * 40 + nt * 16 + ll] = (__bf16)pv[nt][i];
    const __bf16* pp = &Pw[ll * 40 + kb];
    const v16bf ap = ld16(pp, pp + 16);

    #pragma unroll
    for (int t = 0; t < 4; ++t) {
      const __bf16* vp = &Vt[cur][(t * 16 + ll) * 40 + kb2];
      const v16bf bvf = ld16(vp, vp + 16);
      acc[t] = wmma_bf16(ap, bvf, acc[t]);
    }
    if (more) store_kv(cur ^ 1);
  }

  float inv[8];
  #pragma unroll
  for (int i = 0; i < 8; ++i) inv[i] = (lrow[i] > 0.f) ? 1.0f / lrow[i] : 0.f;

  // store ctx as bf16 in [b][s][h*64+d]
  #pragma unroll
  for (int t = 0; t < 4; ++t) {
    #pragma unroll
    for (int i = 0; i < 8; ++i) {
      const float val = acc[t][i] * inv[i];
      const int srow = q0 + i + 8 * lh;
      const int d = t * 16 + ll;
      ctx[((size_t)b * SS + srow) * DM + h * DK + d] = (__bf16)val;
    }
  }
}

// ---------------------------------------------------------------------------
extern "C" void kernel_launch(void* const* d_in, const int* in_sizes, int n_in,
                              void* d_out, int out_size, void* d_ws, size_t ws_size,
                              hipStream_t stream)
{
  (void)in_sizes; (void)n_in; (void)out_size; (void)ws_size;
  const float* x    = (const float*)d_in[0];
  const int*   em   = (const int*)  d_in[1];
  const float* ln1g = (const float*)d_in[2];
  const float* ln1b = (const float*)d_in[3];
  const float* wq   = (const float*)d_in[4];
  const float* bq   = (const float*)d_in[5];
  const float* wk   = (const float*)d_in[6];
  const float* bk   = (const float*)d_in[7];
  const float* wv   = (const float*)d_in[8];
  const float* bv   = (const float*)d_in[9];
  const float* w0   = (const float*)d_in[10];
  const float* b0   = (const float*)d_in[11];
  const float* ln2g = (const float*)d_in[12];
  const float* ln2b = (const float*)d_in[13];
  const float* w1   = (const float*)d_in[14];
  const float* b1   = (const float*)d_in[15];
  const float* w2   = (const float*)d_in[16];
  const float* b2   = (const float*)d_in[17];
  float* out = (float*)d_out;

  char* ws = (char*)d_ws;
  size_t off = 0;
  auto carve = [&](size_t bytes) -> char* {
    char* p = ws + off;
    off += (bytes + 255) & ~(size_t)255;
    return p;
  };
  __bf16* x1   = (__bf16*)carve((size_t)MROWS * DM  * 2);
  __bf16* qb   = (__bf16*)carve((size_t)MROWS * DM  * 2);
  __bf16* kbuf = (__bf16*)carve((size_t)MROWS * DM  * 2);
  __bf16* vbuf = (__bf16*)carve((size_t)MROWS * DM  * 2);
  __bf16* ctx  = (__bf16*)carve((size_t)MROWS * DM  * 2);
  __bf16* x2   = (__bf16*)carve((size_t)MROWS * DM  * 2);
  __bf16* hbuf = (__bf16*)carve((size_t)MROWS * DFF * 2);
  float*  xmid = (float*) carve((size_t)MROWS * DM  * 4);
  __bf16* wqT  = (__bf16*)carve((size_t)DM * DM  * 2);
  __bf16* wkT  = (__bf16*)carve((size_t)DM * DM  * 2);
  __bf16* wvT  = (__bf16*)carve((size_t)DM * DM  * 2);
  __bf16* w0T  = (__bf16*)carve((size_t)DM * DM  * 2);
  __bf16* w1T  = (__bf16*)carve((size_t)DM * DFF * 2);
  __bf16* w2T  = (__bf16*)carve((size_t)DFF * DM * 2);

  const dim3 tb(32, 8);
  transpose_f32_to_bf16<<<dim3(DM / 32,  DM / 32),  tb, 0, stream>>>(wq, wqT, DM,  DM);
  transpose_f32_to_bf16<<<dim3(DM / 32,  DM / 32),  tb, 0, stream>>>(wk, wkT, DM,  DM);
  transpose_f32_to_bf16<<<dim3(DM / 32,  DM / 32),  tb, 0, stream>>>(wv, wvT, DM,  DM);
  transpose_f32_to_bf16<<<dim3(DM / 32,  DM / 32),  tb, 0, stream>>>(w0, w0T, DM,  DM);
  transpose_f32_to_bf16<<<dim3(DFF / 32, DM / 32),  tb, 0, stream>>>(w1, w1T, DM,  DFF);
  transpose_f32_to_bf16<<<dim3(DM / 32,  DFF / 32), tb, 0, stream>>>(w2, w2T, DFF, DM);

  layernorm_to_bf16<<<MROWS, 256, 0, stream>>>(x, ln1g, ln1b, x1);

  const dim3 g1(DM / 256, MROWS / 128);
  gemm_bf16_wmma<<<g1, 256, 0, stream>>>(x1, wqT, bq, nullptr, nullptr, qb,   MROWS, DM, DM, 0);
  gemm_bf16_wmma<<<g1, 256, 0, stream>>>(x1, wkT, bk, nullptr, nullptr, kbuf, MROWS, DM, DM, 0);
  gemm_bf16_wmma<<<g1, 256, 0, stream>>>(x1, wvT, bv, nullptr, nullptr, vbuf, MROWS, DM, DM, 0);

  flash_attn_wmma<<<dim3(SS / 128, BB * NH), 256, 0, stream>>>(qb, kbuf, vbuf, em, ctx);

  gemm_bf16_wmma<<<g1, 256, 0, stream>>>(ctx, w0T, b0, x, xmid, nullptr, MROWS, DM, DM, 1);

  layernorm_to_bf16<<<MROWS, 256, 0, stream>>>(xmid, ln2g, ln2b, x2);

  gemm_bf16_wmma<<<dim3(DFF / 256, MROWS / 128), 256, 0, stream>>>(
      x2, w1T, b1, nullptr, nullptr, hbuf, MROWS, DFF, DM, 2);

  gemm_bf16_wmma<<<g1, 256, 0, stream>>>(hbuf, w2T, b2, xmid, out, nullptr, MROWS, DM, DFF, 1);
}
